// LSTMV1_50740743635507
// MI455X (gfx1250) — matrix operand order repaired and checked by reading, not compiled, not run on hardware
//
#include <hip/hip_runtime.h>
#include <hip/hip_bf16.h>

typedef __attribute__((ext_vector_type(16))) __bf16 v16bf;
typedef __attribute__((ext_vector_type(8)))  float  v8f;
typedef __attribute__((ext_vector_type(4)))  int    v4i;

#define B_   256
#define T_   1024
#define D_   144      // C*Hh = 3*48
#define H_   1024
#define G_   4096     // 4*H
#define XP_  192      // x part padded 144 -> 192 so KP is a multiple of 64
#define KP_  1216     // XP_ + H_  (19 * 64)
#define NT_  (KP_ / 64)   // 19 K-tiles of 64

#define GAS __attribute__((address_space(1)))
#define LAS __attribute__((address_space(3)))

#if defined(__AMDGCN__) && \
    __has_builtin(__builtin_amdgcn_global_load_async_to_lds_b128) && \
    __has_builtin(__builtin_amdgcn_s_wait_asynccnt)
#define USE_ASYNC 1
#else
#define USE_ASYNC 0
#endif

// ---------------------------------------------------------------------------
// Setup: gate-interleaved combined bf16 weights W[g'=4j+gate][0..KP):
//   cols [0,144)   = w_ih[gate*H+j][:]
//   cols [144,192) = 0 (pad)
//   cols [192,KP)  = w_hh[gate*H+j][:]
// bias[g'] = b_ih + b_hh.
// ---------------------------------------------------------------------------
__global__ void lstm_pack_weights(const float* __restrict__ w_ih,
                                  const float* __restrict__ w_hh,
                                  const float* __restrict__ b_ih,
                                  const float* __restrict__ b_hh,
                                  __bf16* __restrict__ Wc,
                                  float* __restrict__ bias) {
  const int gp   = blockIdx.x;
  const int gate = gp & 3;
  const int j    = gp >> 2;
  const int orig = gate * H_ + j;
  for (int k = threadIdx.x; k < KP_; k += blockDim.x) {
    float v;
    if (k < D_)        v = w_ih[orig * D_ + k];
    else if (k < XP_)  v = 0.f;
    else               v = w_hh[orig * H_ + (k - XP_)];
    Wc[(size_t)gp * KP_ + k] = (__bf16)v;
  }
  if (threadIdx.x == 0) bias[gp] = b_ih[orig] + b_hh[orig];
}

// Zero c-state + both activation buffers; fill A0's x-columns with x_{t=0}.
// x(b,t,d) = x0[b*D*T + d*T + t]
__global__ void lstm_init_state(const float* __restrict__ x0,
                                __bf16* __restrict__ A0,
                                __bf16* __restrict__ A1,
                                float* __restrict__ cst) {
  const int i = blockIdx.x * blockDim.x + threadIdx.x;
  if (i < B_ * KP_) {
    const int b = i / KP_, k = i % KP_;
    __bf16 v = (__bf16)0.f;
    if (k < D_) v = (__bf16)x0[(size_t)b * (D_ * T_) + (size_t)k * T_ + 0];
    A0[i] = v;
    A1[i] = (__bf16)0.f;
  }
  if (i < B_ * H_) cst[i] = 0.f;
}

__device__ __forceinline__ float sigmoidf_(float x) {
  return 1.f / (1.f + __expf(-x));
}

// Stage one 64-wide K tile: A (64x64 bf16) + B (128x64 bf16) = 1536 16B chunks,
// exactly 6 per thread (so every wave issues exactly 6 async loads per tile).
__device__ __forceinline__ void stage_tile(const __bf16* __restrict__ Acur,
                                           const __bf16* __restrict__ Wc,
                                           __bf16* Adst, __bf16* Bdst,
                                           int b0, int n0, int k0, int tid) {
#pragma unroll
  for (int q = 0; q < 6; ++q) {
    const __bf16* g;
    __bf16* l;
    if (q < 2) {                    // A chunks: ids [0,512)
      const int id  = tid + q * 256;
      const int row = id >> 3, chk = id & 7;
      g = Acur + (size_t)(b0 + row) * KP_ + k0 + chk * 8;
      l = Adst + row * 64 + chk * 8;
    } else {                        // B chunks: ids [0,1024)
      const int id  = tid + q * 256 - 512;
      const int row = id >> 3, chk = id & 7;
      g = Wc + (size_t)(n0 + row) * KP_ + k0 + chk * 8;
      l = Bdst + row * 64 + chk * 8;
    }
#if USE_ASYNC
    __builtin_amdgcn_global_load_async_to_lds_b128(
        (GAS v4i*)g, (LAS v4i*)l, 0, 0);
#else
    *(uint4*)l = *(const uint4*)g;
#endif
  }
}

// ---------------------------------------------------------------------------
// One fused LSTM timestep:
//   gates(64x128 tile of 256x4096) = A_cur(256xKP bf16) x Wc^T  (WMMA bf16)
//   -> LDS -> sigmoid/tanh -> c,h update -> h (bf16) into A_next
// grid = (32 n-blocks, 4 m-blocks), block = 256 threads = 8 wave32 waves
// laid out 2(M) x 4(N); each wave computes 32x32 (4 WMMA accumulators).
// Weight/activation tiles double-buffered in LDS via async loads.
// ---------------------------------------------------------------------------
__global__ __launch_bounds__(256)
void lstm_step(const float* __restrict__ x0,
               const __bf16* __restrict__ Wc,
               const float* __restrict__ bias,
               float* __restrict__ cst,
               float* __restrict__ hbuf,
               const __bf16* __restrict__ Acur,
               __bf16* __restrict__ Anext,
               int t) {
  __shared__ __bf16 Atile[2 * 64 * 64];    // 16 KB, [buf][m][k]
  __shared__ __bf16 Btile[2 * 128 * 64];   // 32 KB, [buf][n][k]
  __shared__ float  gtile[64 * 132];       // ~34 KB, padded stride

  const int tid  = threadIdx.x;
  const int lane = tid & 31;
  const int wv   = tid >> 5;
  const int bn   = blockIdx.x;             // g' cols [bn*128, ...)
  const int bm   = blockIdx.y;             // batch rows [bm*64, ...)
  const int n0   = bn * 128;
  const int b0   = bm * 64;

  const int m_off = (wv & 1) * 32;         // wave's 32 batch rows
  const int n_off = (wv >> 1) * 32;        // wave's 32 g' cols
  const int half  = lane >> 4;
  const int l16   = lane & 15;

  v8f c00 = {}, c01 = {}, c10 = {}, c11 = {};

#if USE_ASYNC
  stage_tile(Acur, Wc, Atile, Btile, b0, n0, 0, tid);
#endif
  for (int it = 0; it < NT_; ++it) {
    const int cur = it & 1;
#if USE_ASYNC
    if (it + 1 < NT_) {
      stage_tile(Acur, Wc, Atile + (cur ^ 1) * (64 * 64),
                 Btile + (cur ^ 1) * (128 * 64), b0, n0, (it + 1) * 64, tid);
      __builtin_amdgcn_s_wait_asynccnt(6);   // current tile's 6 loads done
    } else {
      __builtin_amdgcn_s_wait_asynccnt(0);
    }
    __syncthreads();
#else
    __syncthreads();
    stage_tile(Acur, Wc, Atile + cur * (64 * 64), Btile + cur * (128 * 64),
               b0, n0, it * 64, tid);
    __syncthreads();
#endif
    const __bf16* At = Atile + cur * (64 * 64);
    const __bf16* Bt = Btile + cur * (128 * 64);

#pragma unroll
    for (int ks = 0; ks < 2; ++ks) {
      const int kb = ks * 32;
      // A fragments (16x32 bf16, ISA 7.12.2): per lane two contiguous
      // 8-element chunks at K = kb + half*8 (+16) -> 2x ds_load_b128.
      v16bf af0, af1;
      {
        const int M0 = m_off + l16;
        const int M1 = m_off + 16 + l16;
#pragma unroll
        for (int v = 0; v < 8; ++v) {
          const int K = kb + ((v < 4) ? 0 : 16) + half * 8 + (v & 3) * 2;
          af0[2 * v]     = At[M0 * 64 + K];
          af0[2 * v + 1] = At[M0 * 64 + K + 1];
          af1[2 * v]     = At[M1 * 64 + K];
          af1[2 * v + 1] = At[M1 * 64 + K + 1];
        }
      }
      // B fragments (32x16 bf16): per lane two contiguous 8-element chunks
      // at K = kb + half*8 (+16) -> 2x ds_load_b128.
      v16bf bf0, bf1;
      {
        const int N0 = n_off + l16;
        const int N1 = n_off + 16 + l16;
#pragma unroll
        for (int v = 0; v < 8; ++v) {
          const int K = kb + ((v < 4) ? 0 : 16) + half * 8 + (v & 3) * 2;
          bf0[2 * v]     = Bt[N0 * 64 + K];
          bf0[2 * v + 1] = Bt[N0 * 64 + K + 1];
          bf1[2 * v]     = Bt[N1 * 64 + K];
          bf1[2 * v + 1] = Bt[N1 * 64 + K + 1];
        }
      }
      c00 = __builtin_amdgcn_wmma_f32_16x16x32_bf16(false, af0, false, bf0,
                                                    (short)0, c00, false, false);
      c01 = __builtin_amdgcn_wmma_f32_16x16x32_bf16(false, af0, false, bf1,
                                                    (short)0, c01, false, false);
      c10 = __builtin_amdgcn_wmma_f32_16x16x32_bf16(false, af1, false, bf0,
                                                    (short)0, c10, false, false);
      c11 = __builtin_amdgcn_wmma_f32_16x16x32_bf16(false, af1, false, bf1,
                                                    (short)0, c11, false, false);
    }
    __syncthreads();
  }

  // spill D tiles to LDS (lane: N=l16; VGPR r: M = 8*half + r)
#pragma unroll
  for (int r = 0; r < 8; ++r) {
    const int Ma = m_off + half * 8 + r;
    const int Mb = m_off + 16 + half * 8 + r;
    gtile[Ma * 132 + n_off + l16]      = c00[r];
    gtile[Ma * 132 + n_off + 16 + l16] = c01[r];
    gtile[Mb * 132 + n_off + l16]      = c10[r];
    gtile[Mb * 132 + n_off + 16 + l16] = c11[r];
  }
  __syncthreads();

  // elementwise LSTM cell: 64 batch x 32 hidden units per block
  for (int p = tid; p < 64 * 32; p += 256) {
    const int bl = p >> 5;          // batch local
    const int jl = p & 31;          // hidden local
    const int gb = n0 + jl * 4;     // interleaved g' base (i,f,g,o)
    const float gi = gtile[bl * 132 + jl * 4 + 0] + bias[gb + 0];
    const float gf = gtile[bl * 132 + jl * 4 + 1] + bias[gb + 1];
    const float gg = gtile[bl * 132 + jl * 4 + 2] + bias[gb + 2];
    const float go = gtile[bl * 132 + jl * 4 + 3] + bias[gb + 3];
    const int b = b0 + bl;
    const int j = bn * 32 + jl;
    const float cold = cst[b * H_ + j];
    const float cnew = sigmoidf_(gf) * cold + sigmoidf_(gi) * tanhf(gg);
    const float h    = sigmoidf_(go) * tanhf(cnew);
    cst[b * H_ + j] = cnew;
    Anext[(size_t)b * KP_ + XP_ + j] = (__bf16)h;   // h-part of next A
    if (t == T_ - 1) hbuf[b * H_ + j] = h;          // for classifier
  }

  // pack x_{t+1} into A_next (writes go to the other buffer -> race-free)
  if (bm == 0 && t + 1 < T_) {
    for (int i = bn * 256 + tid; i < B_ * D_; i += 32 * 256) {
      const int b = i / D_, d = i % D_;
      Anext[(size_t)b * KP_ + d] =
          (__bf16)x0[(size_t)b * (D_ * T_) + (size_t)d * T_ + (t + 1)];
    }
  }
}

// logits = h_last @ w_cls^T + b_cls : 256x10, negligible work
__global__ void lstm_classifier(const float* __restrict__ h,
                                const float* __restrict__ w_cls,
                                const float* __restrict__ b_cls,
                                float* __restrict__ out) {
  __shared__ float red[256];
  const int b = blockIdx.x, tid = threadIdx.x;
  float acc[10];
#pragma unroll
  for (int c = 0; c < 10; ++c) acc[c] = 0.f;
  for (int k = tid; k < H_; k += 256) {
    const float hv = h[b * H_ + k];
#pragma unroll
    for (int c = 0; c < 10; ++c) acc[c] += hv * w_cls[c * H_ + k];
  }
  for (int c = 0; c < 10; ++c) {
    red[tid] = acc[c];
    __syncthreads();
    for (int s = 128; s > 0; s >>= 1) {
      if (tid < s) red[tid] += red[tid + s];
      __syncthreads();
    }
    if (tid == 0) out[b * 10 + c] = red[0] + b_cls[c];
    __syncthreads();
  }
}

extern "C" void kernel_launch(void* const* d_in, const int* in_sizes, int n_in,
                              void* d_out, int out_size, void* d_ws, size_t ws_size,
                              hipStream_t stream) {
  const float* x0    = (const float*)d_in[0];
  const float* w_ih  = (const float*)d_in[1];
  const float* w_hh  = (const float*)d_in[2];
  const float* b_ih  = (const float*)d_in[3];
  const float* b_hh  = (const float*)d_in[4];
  const float* w_cls = (const float*)d_in[5];
  const float* b_cls = (const float*)d_in[6];

  // workspace carve-up (~13.3 MB, all 16B-aligned sizes)
  char* p = (char*)d_ws;
  __bf16* Wc   = (__bf16*)p;  p += (size_t)G_ * KP_ * sizeof(__bf16);
  float*  bias = (float*)p;   p += (size_t)G_ * sizeof(float);
  float*  cst  = (float*)p;   p += (size_t)B_ * H_ * sizeof(float);
  float*  hbuf = (float*)p;   p += (size_t)B_ * H_ * sizeof(float);
  __bf16* A0   = (__bf16*)p;  p += (size_t)B_ * KP_ * sizeof(__bf16);
  __bf16* A1   = (__bf16*)p;

  lstm_pack_weights<<<G_, 256, 0, stream>>>(w_ih, w_hh, b_ih, b_hh, Wc, bias);
  lstm_init_state<<<(B_ * KP_ + 255) / 256, 256, 0, stream>>>(x0, A0, A1, cst);

  for (int t = 0; t < T_; ++t) {
    const __bf16* Ac = (t & 1) ? A1 : A0;
    __bf16*       An = (t & 1) ? A0 : A1;
    lstm_step<<<dim3(32, 4), 256, 0, stream>>>(x0, Wc, bias, cst, hbuf, Ac, An, t);
  }

  lstm_classifier<<<B_, 256, 0, stream>>>(hbuf, w_cls, b_cls, (float*)d_out);
}
